// VGG_50242527429234
// MI455X (gfx1250) — matrix operand order repaired and checked by reading, not compiled, not run on hardware
//
#include <hip/hip_runtime.h>
#include <hip/hip_bf16.h>
#include <stdint.h>

typedef __attribute__((ext_vector_type(16))) _Float16 v16h;
typedef __attribute__((ext_vector_type(8)))  float    v8f;

#define DIV_UP(a,b) (((a)+(b)-1)/(b))

// ---------------- workspace layout (bytes) ----------------
static const size_t OFF_ACT0  = 0;                         // 64*64*64*64*4 = 67,108,864 (largest conv out)
static const size_t OFF_ACT1  = 67108864;                  // 16,777,216 (largest pooled act)
static const size_t OFF_BPACK = OFF_ACT1  + 16777216;      // 75,497,472 (conv1 im2col, exact)
static const size_t OFF_APACK = OFF_BPACK + 75497472;      // 33,554,432 (fw2 pack, exact)
static const size_t OFF_MASK  = OFF_APACK + 33554432;      // 64*512*4 = 131,072
static const size_t OFF_FLAT  = OFF_MASK  + 131072;        // 64*2048*4 = 524,288
static const size_t OFF_FC1   = OFF_FLAT  + 524288;        // 64*4096*4 = 1,048,576
static const size_t OFF_FC2   = OFF_FC1   + 1048576;       // 1,048,576
static const size_t WS_NEEDED = OFF_FC2   + 1048576;       // ~195.7 MB

// ---------------------------------------------------------------------------
// Pack A (weights, row-major [M x K] fp32) into per-lane WMMA f16 A-fragments.
// A 16x32 f16 fragment, this code's contiguous k order: m = lane%16;
//   k = kt*32 + (lane/16)*16 + e   for half index e in 0..15
// Fragment index = (mt*KT + kt)*32 + lane, 16 halfs (32B) each.
// Mtiles may be padded (zero-filled rows) so the GEMM can use 32-row tiles.
// ---------------------------------------------------------------------------
__global__ void pack_a_kernel(const float* __restrict__ W, _Float16* __restrict__ Ap,
                              int M, int K, int Mtiles, int KT) {
  int t = blockIdx.x * blockDim.x + threadIdx.x;
  int total = Mtiles * KT * 32;
  if (t >= total) return;
  int lane = t & 31;
  int kt   = (t >> 5) % KT;
  int mt   = (t >> 5) / KT;
  int m    = mt * 16 + (lane & 15);
  int hi16 = (lane >> 4) << 4;
  v16h frag;
#pragma unroll
  for (int e = 0; e < 16; ++e) {
    int k = kt * 32 + hi16 + e;
    float v = (m < M && k < K) ? W[(size_t)m * K + k] : 0.0f;
    frag[e] = (_Float16)v;
  }
  *(v16h*)(Ap + (size_t)t * 16) = frag;
}

// ---------------------------------------------------------------------------
// Pack B via im2col (3x3 SAME conv) into WMMA f16 B-fragments.
// B 32x16 f16 layout: n = lane%16; k = kt*32 + (lane/16)*16 + e.
// Fragment index = (kt*Ntiles + nt)*32 + lane.  k = ci*9 + (dy*3+dx).
// ---------------------------------------------------------------------------
__global__ void pack_b_im2col_kernel(const float* __restrict__ src, _Float16* __restrict__ Bp,
                                     int C, int H, int W, int K, int KT, int Ntiles) {
  int t = blockIdx.x * blockDim.x + threadIdx.x;
  int total = KT * Ntiles * 32;
  if (t >= total) return;
  int lane = t & 31;
  int nt   = (t >> 5) % Ntiles;
  int kt   = (t >> 5) / Ntiles;
  int n    = nt * 16 + (lane & 15);
  int HW   = H * W;
  int b = n / HW, p = n % HW;
  int h = p / W, w = p % W;
  int kbase = kt * 32 + ((lane >> 4) << 4);
  v16h frag;
#pragma unroll
  for (int e = 0; e < 16; ++e) {
    int k = kbase + e;
    float v = 0.0f;
    if (k < K) {
      int ci = k / 9, r = k % 9;
      int hh = h + r / 3 - 1, ww = w + r % 3 - 1;
      if (hh >= 0 && hh < H && ww >= 0 && ww < W)
        v = src[(((size_t)b * C + ci) * H + hh) * W + ww];
    }
    frag[e] = (_Float16)v;
  }
  *(v16h*)(Bp + (size_t)t * 16) = frag;
}

// ---------------------------------------------------------------------------
// Pack B from row-major activations [N x K] fp32 (FC layers): B[k][n]=src[n*K+k].
// ---------------------------------------------------------------------------
__global__ void pack_b_rowmajor_kernel(const float* __restrict__ src, _Float16* __restrict__ Bp,
                                       int N, int K, int KT, int Ntiles) {
  int t = blockIdx.x * blockDim.x + threadIdx.x;
  int total = KT * Ntiles * 32;
  if (t >= total) return;
  int lane = t & 31;
  int nt   = (t >> 5) % Ntiles;
  int kt   = (t >> 5) / Ntiles;
  int n    = nt * 16 + (lane & 15);
  int kbase = kt * 32 + ((lane >> 4) << 4);
  v16h frag;
#pragma unroll
  for (int e = 0; e < 16; ++e) {
    int k = kbase + e;
    float v = (n < N && k < K) ? src[(size_t)n * K + k] : 0.0f;
    frag[e] = (_Float16)v;
  }
  *(v16h*)(Bp + (size_t)t * 16) = frag;
}

// ---------------------------------------------------------------------------
// Per-accumulator epilogue: C/D 16x16 f32 layout (ISA 7.12.2):
//   VGPR r -> M = r + (lane>=16 ? 8 : 0); N = lane%16.
// ---------------------------------------------------------------------------
__device__ __forceinline__ void epilogue_store(v8f acc, int mbase, int m_off, int n,
                                               const float* __restrict__ bias,
                                               float* __restrict__ out,
                                               int Mlog, int Ntotal, int mode,
                                               int do_relu, int HW, int Cout) {
#pragma unroll
  for (int r = 0; r < 8; ++r) {
    int m = mbase + m_off + r;
    if (m < Mlog && n < Ntotal) {
      float c = acc[r] + bias[m];
      if (do_relu && c < 0.0f) c = 0.0f;
      if (mode == 0) {
        int b = n / HW, p = n % HW;
        out[((size_t)b * Cout + m) * HW + p] = c;
      } else {
        out[(size_t)n * Mlog + m] = c;
      }
    }
  }
}

// ---------------------------------------------------------------------------
// GEMM: one wave computes a 32(M) x 64(N) tile; K-step = 2 A-frag + 4 B-frag
// loads (12x b128) feeding 8x v_wmma_f32_16x16x32_f16 (A reused x4, B x2)
// => ~21 f16-FLOP per operand byte, operands L2-resident.
// K-loop kept at unroll 1: live set ~= 64 acc + 48 fragment + ~12 addr VGPRs,
// well under the 256 directly-addressable VGPRs (no scratch spills).
// mode 0: scatter to NCHW  out[((n/HW)*Cout + m)*HW + n%HW]
// mode 1: row-major        out[n*Mlog + m]
// ---------------------------------------------------------------------------
__global__ void gemm_wmma_kernel(const _Float16* __restrict__ Ap, const _Float16* __restrict__ Bp,
                                 const float* __restrict__ bias, float* __restrict__ out,
                                 int Mpairs, int Ngroups, int KT, int Mlog, int Ntotal,
                                 int mode, int do_relu, int HW, int Cout) {
  int wave = (int)((blockIdx.x * blockDim.x + threadIdx.x) >> 5);
  int lane = threadIdx.x & 31;
  if (wave >= Mpairs * Ngroups) return;      // wave-uniform exit: EXEC all-1s for WMMA
  int ng = wave % Ngroups;
  int mp = wave / Ngroups;
  int NT = Ngroups * 4;

  const v16h* A  = (const v16h*)Ap;
  const v16h* Bv = (const v16h*)Bp;
  v8f acc00 = {}, acc01 = {}, acc02 = {}, acc03 = {};
  v8f acc10 = {}, acc11 = {}, acc12 = {}, acc13 = {};

  size_t abase0 = ((size_t)(mp * 2 + 0) * KT) * 32 + lane;
  size_t abase1 = ((size_t)(mp * 2 + 1) * KT) * 32 + lane;
  size_t bcol   = (size_t)ng * 4 * 32 + lane;

#pragma unroll 1
  for (int kt = 0; kt < KT; ++kt) {
    v16h a0 = A[abase0 + (size_t)kt * 32];
    v16h a1 = A[abase1 + (size_t)kt * 32];
    size_t bb = (size_t)kt * NT * 32 + bcol;
    v16h b0 = Bv[bb];
    v16h b1 = Bv[bb + 32];
    v16h b2 = Bv[bb + 64];
    v16h b3 = Bv[bb + 96];

    acc00 = __builtin_amdgcn_wmma_f32_16x16x32_f16(false, a0, false, b0, (short)0, acc00, false, false);
    acc01 = __builtin_amdgcn_wmma_f32_16x16x32_f16(false, a0, false, b1, (short)0, acc01, false, false);
    acc02 = __builtin_amdgcn_wmma_f32_16x16x32_f16(false, a0, false, b2, (short)0, acc02, false, false);
    acc03 = __builtin_amdgcn_wmma_f32_16x16x32_f16(false, a0, false, b3, (short)0, acc03, false, false);
    acc10 = __builtin_amdgcn_wmma_f32_16x16x32_f16(false, a1, false, b0, (short)0, acc10, false, false);
    acc11 = __builtin_amdgcn_wmma_f32_16x16x32_f16(false, a1, false, b1, (short)0, acc11, false, false);
    acc12 = __builtin_amdgcn_wmma_f32_16x16x32_f16(false, a1, false, b2, (short)0, acc12, false, false);
    acc13 = __builtin_amdgcn_wmma_f32_16x16x32_f16(false, a1, false, b3, (short)0, acc13, false, false);
  }

  int m_off = (lane >> 4) << 3;
  int ncol  = lane & 15;
  int mb0 = (mp * 2 + 0) * 16;
  int mb1 = (mp * 2 + 1) * 16;
  int n0 = (ng * 4 + 0) * 16 + ncol;
  int n1 = (ng * 4 + 1) * 16 + ncol;
  int n2 = (ng * 4 + 2) * 16 + ncol;
  int n3 = (ng * 4 + 3) * 16 + ncol;
  epilogue_store(acc00, mb0, m_off, n0, bias, out, Mlog, Ntotal, mode, do_relu, HW, Cout);
  epilogue_store(acc01, mb0, m_off, n1, bias, out, Mlog, Ntotal, mode, do_relu, HW, Cout);
  epilogue_store(acc02, mb0, m_off, n2, bias, out, Mlog, Ntotal, mode, do_relu, HW, Cout);
  epilogue_store(acc03, mb0, m_off, n3, bias, out, Mlog, Ntotal, mode, do_relu, HW, Cout);
  epilogue_store(acc10, mb1, m_off, n0, bias, out, Mlog, Ntotal, mode, do_relu, HW, Cout);
  epilogue_store(acc11, mb1, m_off, n1, bias, out, Mlog, Ntotal, mode, do_relu, HW, Cout);
  epilogue_store(acc12, mb1, m_off, n2, bias, out, Mlog, Ntotal, mode, do_relu, HW, Cout);
  epilogue_store(acc13, mb1, m_off, n3, bias, out, Mlog, Ntotal, mode, do_relu, HW, Cout);
}

// ---------------- pointwise kernels ----------------
__global__ void maxpool2_kernel(const float* __restrict__ src, float* __restrict__ dst,
                                int total, int Hi, int Wi) {
  int t = blockIdx.x * blockDim.x + threadIdx.x;
  if (t >= total) return;
  int Ho = Hi >> 1, Wo = Wi >> 1;
  int x  = t % Wo;
  int yy = (t / Wo) % Ho;
  int bc = t / (Wo * Ho);
  const float* p = src + ((size_t)bc * Hi + yy * 2) * Wi + x * 2;
  dst[t] = fmaxf(fmaxf(p[0], p[1]), fmaxf(p[Wi], p[Wi + 1]));
}

__global__ void zero_f32_kernel(float* p, int n) {
  int t = blockIdx.x * blockDim.x + threadIdx.x;
  if (t < n) p[t] = 0.0f;
}

__global__ void mask_set_kernel(const int* __restrict__ y, const int* __restrict__ btab,
                                float* __restrict__ mask, int B, int E) {
  int t = blockIdx.x * blockDim.x + threadIdx.x;
  if (t >= B * E) return;
  int b = t / E, j = t % E;
  int c = btab[(size_t)y[b] * E + j];
  if (c >= 0 && c < 512) mask[b * 512 + c] = 1.0f;   // idempotent scatter; sort is unnecessary
}

// h: [64,512,4,4] (post-ReLU conv4b) -> mask, 2x2 maxpool, flatten to [64,2048]
__global__ void mask_pool_flatten_kernel(const float* __restrict__ h, const float* __restrict__ mask,
                                         float* __restrict__ flat) {
  int t = blockIdx.x * blockDim.x + threadIdx.x;
  if (t >= 64 * 512 * 4) return;
  int px = t & 1, py = (t >> 1) & 1;
  int c  = (t >> 2) % 512;
  int b  = (t >> 2) / 512;
  const float* base = h + (((size_t)b * 512 + c) * 4 + py * 2) * 4 + px * 2;
  float v = fmaxf(fmaxf(base[0], base[1]), fmaxf(base[4], base[5]));
  v *= mask[b * 512 + c];
  flat[(size_t)b * 2048 + c * 4 + py * 2 + px] = v;
}

// ---------------- host-side layer drivers ----------------
static void run_conv(const float* src, int B, int C, int H, int W,
                     const float* wgt, const float* bias, int O,
                     float* dst, char* ws, hipStream_t stream) {
  int K = C * 9, KT = DIV_UP(K, 32);
  int N = B * H * W, Ntiles = N / 16, Ngroups = Ntiles / 4;
  int MtilesPad = 2 * DIV_UP(DIV_UP(O, 16), 2);     // even # of 16-row tiles (zero-padded)
  int Mpairs = MtilesPad / 2;
  _Float16* Ap = (_Float16*)(ws + OFF_APACK);
  _Float16* Bp = (_Float16*)(ws + OFF_BPACK);
  { int tot = MtilesPad * KT * 32;
    pack_a_kernel<<<DIV_UP(tot, 256), 256, 0, stream>>>(wgt, Ap, O, K, MtilesPad, KT); }
  { int tot = KT * Ntiles * 32;
    pack_b_im2col_kernel<<<DIV_UP(tot, 256), 256, 0, stream>>>(src, Bp, C, H, W, K, KT, Ntiles); }
  { int thr = Mpairs * Ngroups * 32;
    gemm_wmma_kernel<<<DIV_UP(thr, 256), 256, 0, stream>>>(Ap, Bp, bias, dst,
        Mpairs, Ngroups, KT, O, N, /*mode=*/0, /*relu=*/1, H * W, O); }
}

static void run_fc(const float* src, int Nb, int K,
                   const float* wgt, const float* bias, int O,
                   float* dst, int do_relu, char* ws, hipStream_t stream) {
  int KT = DIV_UP(K, 32);
  int Ntiles = DIV_UP(Nb, 16);           // 64 -> 4 tiles -> exactly one 64-col group
  int Ngroups = DIV_UP(Ntiles, 4);
  int MtilesPad = 2 * DIV_UP(DIV_UP(O, 16), 2);
  int Mpairs = MtilesPad / 2;
  _Float16* Ap = (_Float16*)(ws + OFF_APACK);
  _Float16* Bp = (_Float16*)(ws + OFF_BPACK);
  { int tot = MtilesPad * KT * 32;
    pack_a_kernel<<<DIV_UP(tot, 256), 256, 0, stream>>>(wgt, Ap, O, K, MtilesPad, KT); }
  { int tot = KT * Ntiles * 32;
    pack_b_rowmajor_kernel<<<DIV_UP(tot, 256), 256, 0, stream>>>(src, Bp, Nb, K, KT, Ntiles); }
  { int thr = Mpairs * Ngroups * 32;
    gemm_wmma_kernel<<<DIV_UP(thr, 256), 256, 0, stream>>>(Ap, Bp, bias, dst,
        Mpairs, Ngroups, KT, O, Nb, /*mode=*/1, do_relu, 1, O); }
}

extern "C" void kernel_launch(void* const* d_in, const int* in_sizes, int n_in,
                              void* d_out, int out_size, void* d_ws, size_t ws_size,
                              hipStream_t stream) {
  if (ws_size < WS_NEEDED) return;
  const float* x    = (const float*)d_in[0];
  const int*   y    = (const int*)  d_in[1];
  const int*   btab = (const int*)  d_in[2];
  const float* cw0  = (const float*)d_in[3];  const float* cb0  = (const float*)d_in[4];
  const float* cw1  = (const float*)d_in[5];  const float* cb1  = (const float*)d_in[6];
  const float* cw2a = (const float*)d_in[7];  const float* cb2a = (const float*)d_in[8];
  const float* cw2b = (const float*)d_in[9];  const float* cb2b = (const float*)d_in[10];
  const float* cw3a = (const float*)d_in[11]; const float* cb3a = (const float*)d_in[12];
  const float* cw3b = (const float*)d_in[13]; const float* cb3b = (const float*)d_in[14];
  const float* cw4a = (const float*)d_in[15]; const float* cb4a = (const float*)d_in[16];
  const float* cw4b = (const float*)d_in[17]; const float* cb4b = (const float*)d_in[18];
  const float* fw1  = (const float*)d_in[19]; const float* fb1  = (const float*)d_in[20];
  const float* fw2  = (const float*)d_in[21]; const float* fb2  = (const float*)d_in[22];
  const float* fw3  = (const float*)d_in[23]; const float* fb3  = (const float*)d_in[24];
  (void)in_sizes; (void)n_in; (void)out_size;

  char*  ws   = (char*)d_ws;
  float* ACT0 = (float*)(ws + OFF_ACT0);
  float* ACT1 = (float*)(ws + OFF_ACT1);
  float* MASK = (float*)(ws + OFF_MASK);
  float* FLAT = (float*)(ws + OFF_FLAT);
  float* FC1  = (float*)(ws + OFF_FC1);
  float* FC2  = (float*)(ws + OFF_FC2);
  const int B = 64;

  // block0: conv0 + relu -> pool (64x64 -> 32x32)
  run_conv(x, B, 3, 64, 64, cw0, cb0, 64, ACT0, ws, stream);
  { int tot = B * 64 * 32 * 32;
    maxpool2_kernel<<<DIV_UP(tot, 256), 256, 0, stream>>>(ACT0, ACT1, tot, 64, 64); }

  // block1: conv1 + relu -> pool (32x32 -> 16x16)
  run_conv(ACT1, B, 64, 32, 32, cw1, cb1, 128, ACT0, ws, stream);
  { int tot = B * 128 * 16 * 16;
    maxpool2_kernel<<<DIV_UP(tot, 256), 256, 0, stream>>>(ACT0, ACT1, tot, 32, 32); }

  // block2: conv2a+relu, conv2b+relu -> pool (16x16 -> 8x8)
  run_conv(ACT1, B, 128, 16, 16, cw2a, cb2a, 256, ACT0, ws, stream);
  run_conv(ACT0, B, 256, 16, 16, cw2b, cb2b, 256, ACT0, ws, stream);  // src consumed by packB first
  { int tot = B * 256 * 8 * 8;
    maxpool2_kernel<<<DIV_UP(tot, 256), 256, 0, stream>>>(ACT0, ACT1, tot, 16, 16); }

  // block3: conv3a+relu, conv3b+relu -> pool (8x8 -> 4x4)
  run_conv(ACT1, B, 256, 8, 8, cw3a, cb3a, 512, ACT0, ws, stream);
  run_conv(ACT0, B, 512, 8, 8, cw3b, cb3b, 512, ACT0, ws, stream);
  { int tot = B * 512 * 4 * 4;
    maxpool2_kernel<<<DIV_UP(tot, 256), 256, 0, stream>>>(ACT0, ACT1, tot, 8, 8); }

  // block4: conv4a+relu, conv4b+relu -> [64,512,4,4]
  run_conv(ACT1, B, 512, 4, 4, cw4a, cb4a, 512, ACT0, ws, stream);
  run_conv(ACT0, B, 512, 4, 4, cw4b, cb4b, 512, ACT0, ws, stream);

  // engram mask + pool (4x4 -> 2x2) + flatten -> [64, 2048]
  { int n = B * 512;
    zero_f32_kernel<<<DIV_UP(n, 256), 256, 0, stream>>>(MASK, n); }
  { int n = B * 128;
    mask_set_kernel<<<DIV_UP(n, 256), 256, 0, stream>>>(y, btab, MASK, B, 128); }
  { int n = B * 512 * 4;
    mask_pool_flatten_kernel<<<DIV_UP(n, 256), 256, 0, stream>>>(ACT0, MASK, FLAT); }

  // classifier
  run_fc(FLAT, B, 2048, fw1, fb1, 4096, FC1, /*relu=*/1, ws, stream);
  run_fc(FC1,  B, 4096, fw2, fb2, 4096, FC2, /*relu=*/1, ws, stream);
  run_fc(FC2,  B, 4096, fw3, fb3, 1000, (float*)d_out, /*relu=*/0, ws, stream);
}
